// MultiHeadAttentionBlock_54511724921077
// MI455X (gfx1250) — compile-verified, matrix operand order checked by reading, not run on hardware
//
#include <hip/hip_runtime.h>

// ---------------------------------------------------------------------------
// MHA block for MI455X (gfx1250, wave32, WMMA).
// Precision strategy: bf16 operands, f32 accumulation via v_wmma_f32_16x16x32_bf16.
// Flash-attention schedule: S x S scores never materialized in HBM.
// ---------------------------------------------------------------------------

typedef __attribute__((ext_vector_type(16))) __bf16 v16bf;
typedef __attribute__((ext_vector_type(8)))  float  v8f;

#define D_MODEL 1024
#define SEQ     2048
#define NB      2
#define NH      16
#define HD      64
#define MTOT    (NB * SEQ)   // 4096 rows of activations

// ---- helpers ---------------------------------------------------------------

__device__ __forceinline__ unsigned short f2bf_bits(float f) {
  union { float f; unsigned int u; } x; x.f = f;
  unsigned int r = x.u + 0x7FFFu + ((x.u >> 16) & 1u);   // round-to-nearest-even
  return (unsigned short)(r >> 16);
}
__device__ __forceinline__ __bf16 f2bf(float f) {
  union { unsigned short u; __bf16 b; } c; c.u = f2bf_bits(f);
  return c.b;
}

union FragBF {
  v16bf v;
  uint4 q[2];
};

// A-style 16x32 bf16 fragment load from a row-major matrix (row stride ld).
// ISA layout (16-bit A-matrix 16x32): lanes 0-15 hold rows 0-15 with
// K = {0..7, 16..23} in elements {0..7, 8..15}; lanes 16-31 hold the same rows
// with K = {8..15, 24..31}.  Two contiguous 16B loads per lane.
__device__ __forceinline__ v16bf load_frag_rm(const __bf16* __restrict__ base,
                                              int row0, int ld, int k0, int lane) {
  const __bf16* p = base + (size_t)(row0 + (lane & 15)) * ld
                         + (size_t)(k0 + ((lane >> 4) << 3));
  FragBF f;
  f.q[0] = *(const uint4*)(p);
  f.q[1] = *(const uint4*)(p + 16);
  return f.v;
}

__device__ __forceinline__ v8f wmma_bf16(v16bf a, v16bf b, v8f c) {
  return __builtin_amdgcn_wmma_f32_16x16x32_bf16(
      /*neg_a=*/false, a, /*neg_b=*/false, b,
      /*c_mod=*/(short)0, c, /*reuse_a=*/false, /*reuse_b=*/false);
}

// ---- kernel 0: fp32 -> bf16 conversion ------------------------------------

__global__ void cvt_f32_bf16_kernel(const float* __restrict__ in,
                                    __bf16* __restrict__ out, int n) {
  for (int i = blockIdx.x * blockDim.x + threadIdx.x; i < n;
       i += gridDim.x * blockDim.x)
    out[i] = f2bf(in[i]);
}

// ---- kernel 0b: V [B,S,H,Dh] -> VT [B,H,Dh,S] ------------------------------

__global__ void transpose_v_kernel(const __bf16* __restrict__ v,
                                   __bf16* __restrict__ vt, int total) {
  for (int o = blockIdx.x * blockDim.x + threadIdx.x; o < total;
       o += gridDim.x * blockDim.x) {
    int s = o & (SEQ - 1);
    int d = (o >> 11) & (HD - 1);
    int h = (o >> 17) & (NH - 1);
    int b = o >> 21;
    vt[o] = v[(size_t)(b * SEQ + s) * D_MODEL + h * HD + d];
  }
}

// ---- kernel 1: C = A[MxK] @ W[NxK]^T + bias --------------------------------
// Wave tile 32x64 (2x4 WMMA accumulators), block = 8 waves = 128x128 tile.
// B-fragment of W^T is an A-style load of W (both row-major over K).

template <int OUT_BF16>
__global__ __launch_bounds__(256) void gemm_xwt_kernel(
    const __bf16* __restrict__ A, const __bf16* __restrict__ W,
    const float* __restrict__ bias,
    float* __restrict__ outF, __bf16* __restrict__ outB,
    int M, int N, int K) {
  const int lane = threadIdx.x & 31;
  const int wid  = threadIdx.x >> 5;
  const int m0 = blockIdx.y * 128 + (wid & 3) * 32;
  const int n0 = blockIdx.x * 128 + (wid >> 2) * 64;

  v8f acc[2][4] = {};

  for (int k0 = 0; k0 < K; k0 += 32) {
    v16bf a0 = load_frag_rm(A, m0,      K, k0, lane);
    v16bf a1 = load_frag_rm(A, m0 + 16, K, k0, lane);
    v16bf b0 = load_frag_rm(W, n0,      K, k0, lane);
    v16bf b1 = load_frag_rm(W, n0 + 16, K, k0, lane);
    v16bf b2 = load_frag_rm(W, n0 + 32, K, k0, lane);
    v16bf b3 = load_frag_rm(W, n0 + 48, K, k0, lane);
    acc[0][0] = wmma_bf16(a0, b0, acc[0][0]);
    acc[0][1] = wmma_bf16(a0, b1, acc[0][1]);
    acc[0][2] = wmma_bf16(a0, b2, acc[0][2]);
    acc[0][3] = wmma_bf16(a0, b3, acc[0][3]);
    acc[1][0] = wmma_bf16(a1, b0, acc[1][0]);
    acc[1][1] = wmma_bf16(a1, b1, acc[1][1]);
    acc[1][2] = wmma_bf16(a1, b2, acc[1][2]);
    acc[1][3] = wmma_bf16(a1, b3, acc[1][3]);
  }

  // C/D layout: VGPR e, lane l -> row = e + 8*(l>=16), col = l&15.
#pragma unroll
  for (int mi = 0; mi < 2; ++mi)
#pragma unroll
    for (int ni = 0; ni < 4; ++ni) {
      const int c = n0 + ni * 16 + (lane & 15);
      const float bv = bias[c];
#pragma unroll
      for (int e = 0; e < 8; ++e) {
        const int r = m0 + mi * 16 + e + ((lane >> 4) << 3);
        const float val = acc[mi][ni][e] + bv;
        if (OUT_BF16) outB[(size_t)r * N + c] = f2bf(val);
        else          outF[(size_t)r * N + c] = val;
      }
    }
}

// ---- kernel 2: causal flash attention --------------------------------------
// One wave per 16-row Q block; j-tiles of 32; online softmax; P staged through
// 1KB wave-private LDS to re-shape C-layout f32 -> A-layout bf16 fragment.

__global__ __launch_bounds__(256) void flash_attn_kernel(
    const __bf16* __restrict__ Q, const __bf16* __restrict__ Kc,
    const __bf16* __restrict__ VT, __bf16* __restrict__ ctx) {
  __shared__ __align__(16) __bf16 lds_p[8][16 * 32];

  const int lane = threadIdx.x & 31;
  const int wid  = threadIdx.x >> 5;
  const int bh = blockIdx.x;          // b*NH + h
  const int b  = bh >> 4;
  const int h  = bh & 15;
  const int q0 = blockIdx.y * 128 + wid * 16;   // row within S (wave-uniform)
  const int hi = lane >> 4;           // 0/1 half select
  const int cl = lane & 15;

  const __bf16* qb = Q  + (size_t)b * SEQ * D_MODEL + (size_t)h * HD;
  const __bf16* kb = Kc + (size_t)b * SEQ * D_MODEL + (size_t)h * HD;
  const __bf16* vb = VT + (size_t)bh * HD * SEQ;

  // Whole 16x64 Q tile as two A-fragments, kept in registers.
  v16bf qa0 = load_frag_rm(qb, q0, D_MODEL, 0,  lane);
  v16bf qa1 = load_frag_rm(qb, q0, D_MODEL, 32, lane);

  v8f o[4] = {};
  float mst[8], lst[8];
#pragma unroll
  for (int e = 0; e < 8; ++e) { mst[e] = -3.0e38f; lst[e] = 0.f; }

  const float scale = 0.125f;  // 1/sqrt(64)
  __bf16* lp = &lds_p[wid][0];

  for (int j0 = 0; j0 < q0 + 16; j0 += 32) {
    // S(16x32) = Q(16x64) @ K^T : B-fragment = A-style load of K rows.
    v8f s0 = {}, s1 = {};
    {
      v16bf k00 = load_frag_rm(kb, j0,      D_MODEL, 0,  lane);
      v16bf k01 = load_frag_rm(kb, j0,      D_MODEL, 32, lane);
      v16bf k10 = load_frag_rm(kb, j0 + 16, D_MODEL, 0,  lane);
      v16bf k11 = load_frag_rm(kb, j0 + 16, D_MODEL, 32, lane);
      s0 = wmma_bf16(qa0, k00, s0);
      s0 = wmma_bf16(qa1, k01, s0);
      s1 = wmma_bf16(qa0, k10, s1);
      s1 = wmma_bf16(qa1, k11, s1);
    }

    // Online softmax per row (rows live across 16-lane halves at fixed elem e).
#pragma unroll
    for (int e = 0; e < 8; ++e) {
      const int i = q0 + e + (hi << 3);
      float x0 = s0[e] * scale;
      float x1 = s1[e] * scale;
      if (j0 + cl      > i) x0 = -3.0e38f;   // causal mask
      if (j0 + 16 + cl > i) x1 = -3.0e38f;
      float mloc = fmaxf(x0, x1);
#pragma unroll
      for (int off = 1; off < 16; off <<= 1)
        mloc = fmaxf(mloc, __shfl_xor(mloc, off, 32));
      const float mnew  = fmaxf(mst[e], mloc);
      const float alpha = __expf(mst[e] - mnew);
      mst[e] = mnew;
      const float p0 = __expf(x0 - mnew);
      const float p1 = __expf(x1 - mnew);
      float rs = p0 + p1;
#pragma unroll
      for (int off = 1; off < 16; off <<= 1)
        rs += __shfl_xor(rs, off, 32);
      lst[e] = lst[e] * alpha + rs;
      o[0][e] *= alpha; o[1][e] *= alpha; o[2][e] *= alpha; o[3][e] *= alpha;
      const int r = e + (hi << 3);
      lp[r * 32 + cl]      = f2bf(p0);       // P tile row-major 16x32 in LDS
      lp[r * 32 + 16 + cl] = f2bf(p1);
    }

    // LDS writes above feed A-fragment reads below (cross-lane within wave).
    __asm__ volatile("s_wait_dscnt 0" ::: "memory");

    FragBF pf;
    const __bf16* pp = lp + cl * 32 + (hi << 3);
    pf.q[0] = *(const uint4*)(pp);
    pf.q[1] = *(const uint4*)(pp + 16);

    // O(16x64) += P(16x32) @ V(32x64): B-fragment = A-style load of VT rows.
#pragma unroll
    for (int n = 0; n < 4; ++n) {
      v16bf vf = load_frag_rm(vb, n * 16, SEQ, j0, lane);
      o[n] = wmma_bf16(pf.v, vf, o[n]);
    }
    __asm__ volatile("" ::: "memory");  // keep next iter's LDS stores after reads
  }

  // Normalize and store context in [B*S, D_MODEL] bf16 for the out projection.
#pragma unroll
  for (int n = 0; n < 4; ++n) {
    const int c = h * HD + n * 16 + cl;
#pragma unroll
    for (int e = 0; e < 8; ++e) {
      const int r = b * SEQ + q0 + e + (hi << 3);
      ctx[(size_t)r * D_MODEL + c] = f2bf(o[n][e] / lst[e]);
    }
  }
}

// ---- host launch -----------------------------------------------------------

extern "C" void kernel_launch(void* const* d_in, const int* in_sizes, int n_in,
                              void* d_out, int out_size, void* d_ws, size_t ws_size,
                              hipStream_t stream) {
  const float* q_f = (const float*)d_in[0];
  const float* k_f = (const float*)d_in[1];
  const float* v_f = (const float*)d_in[2];
  // d_in[3] = causal tril mask: reproduced analytically in flash_attn_kernel.
  const float* Wq = (const float*)d_in[4];
  const float* bq = (const float*)d_in[5];
  const float* Wk = (const float*)d_in[6];
  const float* bk = (const float*)d_in[7];
  const float* Wv = (const float*)d_in[8];
  const float* bv = (const float*)d_in[9];
  const float* Wo = (const float*)d_in[10];
  const float* bo = (const float*)d_in[11];

  char* ws = (char*)d_ws;
  size_t off = 0;
  auto alloc = [&](size_t elems) {
    __bf16* p = (__bf16*)(ws + off);
    off += ((elems * sizeof(__bf16)) + 255) & ~(size_t)255;
    return p;
  };

  const size_t nx = (size_t)MTOT * D_MODEL;      // 4M activation elems
  const size_t nw = (size_t)D_MODEL * D_MODEL;   // 1M weight elems

  __bf16* qin = alloc(nx);
  __bf16* kin = alloc(nx);
  __bf16* vin = alloc(nx);
  __bf16* wqb = alloc(nw);
  __bf16* wkb = alloc(nw);
  __bf16* wvb = alloc(nw);
  __bf16* wob = alloc(nw);
  __bf16* qp  = alloc(nx);   // post-projection Q
  __bf16* kp  = alloc(nx);   // post-projection K
  __bf16* vp  = alloc(nx);   // post-projection V
  __bf16* vt  = alloc(nx);   // V transposed [B,H,Dh,S]
  __bf16* ctx = alloc(nx);   // attention context
  (void)ws_size; (void)in_sizes; (void)n_in; (void)out_size;

  // fp32 -> bf16
  cvt_f32_bf16_kernel<<<1024, 256, 0, stream>>>(q_f, qin, (int)nx);
  cvt_f32_bf16_kernel<<<1024, 256, 0, stream>>>(k_f, kin, (int)nx);
  cvt_f32_bf16_kernel<<<1024, 256, 0, stream>>>(v_f, vin, (int)nx);
  cvt_f32_bf16_kernel<<<512,  256, 0, stream>>>(Wq, wqb, (int)nw);
  cvt_f32_bf16_kernel<<<512,  256, 0, stream>>>(Wk, wkb, (int)nw);
  cvt_f32_bf16_kernel<<<512,  256, 0, stream>>>(Wv, wvb, (int)nw);
  cvt_f32_bf16_kernel<<<512,  256, 0, stream>>>(Wo, wob, (int)nw);

  // projections: [4096x1024] @ [1024x1024]^T + bias -> bf16
  dim3 ggrd(D_MODEL / 128, MTOT / 128);   // (8, 32)
  gemm_xwt_kernel<1><<<ggrd, 256, 0, stream>>>(qin, wqb, bq, nullptr, qp,
                                               MTOT, D_MODEL, D_MODEL);
  gemm_xwt_kernel<1><<<ggrd, 256, 0, stream>>>(kin, wkb, bk, nullptr, kp,
                                               MTOT, D_MODEL, D_MODEL);
  gemm_xwt_kernel<1><<<ggrd, 256, 0, stream>>>(vin, wvb, bv, nullptr, vp,
                                               MTOT, D_MODEL, D_MODEL);

  // V -> V^T per head for contiguous P@V B-fragments
  transpose_v_kernel<<<1024, 256, 0, stream>>>(vp, vt, (int)nx);

  // causal flash attention: grid (B*H, S/128), 8 waves per block
  flash_attn_kernel<<<dim3(NB * NH, SEQ / 128), 256, 0, stream>>>(qp, kp, vt, ctx);

  // output projection -> fp32 d_out
  gemm_xwt_kernel<0><<<ggrd, 256, 0, stream>>>(ctx, wob, bo, (float*)d_out,
                                               nullptr, MTOT, D_MODEL, D_MODEL);
}
